// BehaviorSpecificPFF_41059887350445
// MI455X (gfx1250) — matrix operand order repaired and checked by reading, not compiled, not run on hardware
//
#include <hip/hip_runtime.h>

// Problem constants (from reference): B=32, T=512, H=512, F=2048, N_B=4
#define BATCH 32
#define TT    512
#define HH    512
#define FF    2048
#define NB    4
#define NTOK  (BATCH*TT)       // 16384 tokens
#define MAXTILES (NTOK/16)     // 1024 16-token tiles per behavior (worst case)

typedef __attribute__((ext_vector_type(16))) __bf16 v16bf;
typedef __attribute__((ext_vector_type(8)))  __bf16 v8bf;
typedef __attribute__((ext_vector_type(8)))  float  v8f;
typedef __attribute__((__vector_size__(16))) int    v4i;

#ifndef __has_builtin
#define __has_builtin(x) 0
#endif

#if __has_builtin(__builtin_amdgcn_global_load_async_to_lds_b128)
#define HAS_ASYNC_LDS 1
#else
#define HAS_ASYNC_LDS 0
#endif

// 16-byte global->LDS copy: async (ASYNCcnt-tracked) when available.
__device__ __forceinline__ void cp16(void* l, const void* g) {
#if HAS_ASYNC_LDS
  __builtin_amdgcn_global_load_async_to_lds_b128(
      (__attribute__((address_space(1))) v4i*)(v4i*)g,
      (__attribute__((address_space(3))) v4i*)(v4i*)l, 0, 0);
#else
  *(uint4*)l = *(const uint4*)g;
#endif
}

__device__ __forceinline__ void wait_async8() {
#if HAS_ASYNC_LDS
#if __has_builtin(__builtin_amdgcn_s_wait_asynccnt)
  __builtin_amdgcn_s_wait_asynccnt(8);
#else
  asm volatile("s_wait_asynccnt 0x8" ::: "memory");
#endif
#endif
}
__device__ __forceinline__ void wait_async0() {
#if HAS_ASYNC_LDS
#if __has_builtin(__builtin_amdgcn_s_wait_asynccnt)
  __builtin_amdgcn_s_wait_asynccnt(0);
#else
  asm volatile("s_wait_asynccnt 0x0" ::: "memory");
#endif
#endif
}

// ---- fragment loaders (CDNA5 WMMA 16-bit layouts, ISA 7.12.2) -------------
__device__ __forceinline__ v16bf ld_frag_a(const __bf16* p) {
  v8bf lo = *(const v8bf*)(p);        // K base+0..7
  v8bf hi = *(const v8bf*)(p + 16);   // K base+16..23
  v16bf r;
#pragma unroll
  for (int i = 0; i < 8; ++i) { r[i] = lo[i]; r[i + 8] = hi[i]; }
  return r;
}
__device__ __forceinline__ v16bf ld_frag_b(const __bf16* p) {
  v8bf lo = *(const v8bf*)(p);        // K base+0..7
  v8bf hi = *(const v8bf*)(p + 16);   // K base+16..23
  v16bf r;
#pragma unroll
  for (int i = 0; i < 8; ++i) { r[i] = lo[i]; r[i + 8] = hi[i]; }
  return r;
}

// ---- kernel 1: zero output + counters -------------------------------------
__global__ void zero_kernel(float* __restrict__ out, int n, int* __restrict__ cnt) {
  int i = blockIdx.x * blockDim.x + threadIdx.x;
  if (4 * i + 3 < n) {
    float4 z; z.x = 0.f; z.y = 0.f; z.z = 0.f; z.w = 0.f;
    ((float4*)out)[i] = z;
  }
  if (blockIdx.x == 0 && threadIdx.x < 8) cnt[threadIdx.x] = 0;
}

// ---- kernel 2: bucket tokens by behavior ----------------------------------
__global__ void bucket_kernel(const int* __restrict__ bseq, int* __restrict__ cnt,
                              int* __restrict__ idx) {
  int t = blockIdx.x * blockDim.x + threadIdx.x;
  if (t >= NTOK) return;
  int v = bseq[t];
  if (v > 0) {
    int pos = atomicAdd(&cnt[v - 1], 1);
    idx[(v - 1) * NTOK + pos] = t;
  }
}

// ---- kernel 3: f32 -> bf16 flat convert (x) -------------------------------
__global__ void cvt_kernel(const float* __restrict__ in, __bf16* __restrict__ out,
                           int n) {
  int i = blockIdx.x * blockDim.x + threadIdx.x;
  if (4 * i + 3 < n) {
    const float4 v = ((const float4*)in)[i];
    out[4 * i + 0] = (__bf16)v.x;
    out[4 * i + 1] = (__bf16)v.y;
    out[4 * i + 2] = (__bf16)v.z;
    out[4 * i + 3] = (__bf16)v.w;
  }
}

// ---- kernel 4: f32 [R][C] -> bf16 transposed [C][R], per behavior (grid.z) -
__global__ void transpose_cvt_kernel(const float* __restrict__ in,
                                     __bf16* __restrict__ outT, int R, int C) {
  __shared__ float t[32][33];
  const float* inb = in + (size_t)blockIdx.z * R * C;
  __bf16* outb = outT + (size_t)blockIdx.z * R * C;
  int c0 = blockIdx.x * 32, r0 = blockIdx.y * 32;
  int tx = threadIdx.x, ty = threadIdx.y;  // 32 x 8
#pragma unroll
  for (int i = 0; i < 32; i += 8)
    t[ty + i][tx] = inb[(size_t)(r0 + ty + i) * C + c0 + tx];
  __syncthreads();
#pragma unroll
  for (int i = 0; i < 32; i += 8)
    outb[(size_t)(c0 + ty + i) * R + r0 + tx] = (__bf16)t[tx][ty + i];
}

// ---- kernel 5: fused grouped FFN via bf16 WMMA, async double-buffered -----
__global__ __launch_bounds__(256) void ffn_kernel(
    const __bf16* __restrict__ xb,    // [NTOK][H]  bf16
    const __bf16* __restrict__ w1t,   // [NB*F][H]  bf16 (K-major for GEMM1 B)
    const __bf16* __restrict__ w2t,   // [NB*H][F]  bf16 (K-major for GEMM2 B)
    const float* __restrict__ B1, const float* __restrict__ B2,
    float* __restrict__ out,
    const int* __restrict__ cnt, const int* __restrict__ idx) {
  __shared__ alignas(16) __bf16 xs[16 * 512];        // gathered x tile   16KB
  __shared__ alignas(16) __bf16 hs[16 * 512];        // relu intermediate 16KB
  __shared__ alignas(16) __bf16 wst[2][512 * 32];    // wgt slab ping-pong 64KB

  const int n    = blockIdx.x / MAXTILES;
  const int tile = blockIdx.x % MAXTILES;
  const int cn = cnt[n];
  if (tile * 16 >= cn) return;  // uniform per block
  const int rows = (cn - tile * 16 < 16) ? (cn - tile * 16) : 16;

  const int tid  = threadIdx.x;
  const int lane = tid & 31;
  const int wave = tid >> 5;
  const int lrow = lane & 15;
  const int hi16 = (lane & 16) ? 1 : 0;
  const int* tl = idx + n * NTOK + tile * 16;

  int fbase = 0;

  // stage W1T slab [fbase+col][k0:k0+32] -> wst[buf][col][k], 2048 x 16B chunks
  auto stage_w1 = [&](int buf, int k0) {
#pragma unroll
    for (int i = 0; i < 8; ++i) {
      int c = tid + i * 256;
      int row = c >> 2, part = c & 3;
      const __bf16* g =
          w1t + (size_t)(n * FF + fbase + row) * HH + k0 + part * 8;
      cp16((char*)&wst[buf][0] + c * 16, g);
    }
  };
  // stage W2T slab [col][fbase+k0 : +32] -> wst[buf][col][k]
  auto stage_w2 = [&](int buf, int k0) {
#pragma unroll
    for (int i = 0; i < 8; ++i) {
      int c = tid + i * 256;
      int row = c >> 2, part = c & 3;
      const __bf16* g =
          w2t + (size_t)(n * HH + row) * FF + fbase + k0 + part * 8;
      cp16((char*)&wst[buf][0] + c * 16, g);
    }
  };

  // ---- async gather of the 16-token x tile (1024 x 16B chunks) ----
#pragma unroll
  for (int i = 0; i < 4; ++i) {
    int c = tid + i * 256;
    int row = c >> 6;
    int rc = (row < rows) ? row : 0;
    int tok = tl[rc];
    cp16((char*)xs + c * 16, xb + (size_t)tok * HH + (c & 63) * 8);
  }

  const v8f vzero = {0.f, 0.f, 0.f, 0.f, 0.f, 0.f, 0.f, 0.f};
  v8f acc2[4];
#pragma unroll
  for (int j = 0; j < 4; ++j) acc2[j] = vzero;

  for (int fc = 0; fc < 4; ++fc) {  // F in 4 chunks of 512
    fbase = fc * 512;
    v8f acc1[4];
#pragma unroll
    for (int j = 0; j < 4; ++j) acc1[j] = vzero;

    // ---- GEMM1: h_chunk = x @ W1[:, fbase:fbase+512] ----
    {
      int buf = 0;
      stage_w1(0, 0);
      for (int k0 = 0; k0 < HH; k0 += 32) {
        if (k0 + 32 < HH) { stage_w1(buf ^ 1, k0 + 32); wait_async8(); }
        else              { wait_async0(); }
        __syncthreads();
        v16bf a = ld_frag_a(&xs[lrow * 512 + k0 + hi16 * 8]);
#pragma unroll
        for (int j = 0; j < 4; ++j) {
          int col = (wave * 4 + j) * 16 + lrow;
          v16bf b = ld_frag_b(&wst[buf][col * 32 + hi16 * 8]);
          acc1[j] = __builtin_amdgcn_wmma_f32_16x16x32_bf16(
              false, a, false, b, (short)0, acc1[j], false, false);
        }
        __syncthreads();  // protect this buf before it is re-staged
        buf ^= 1;
      }
    }

    // ---- bias + ReLU -> hs (bf16), C/D lane layout ----
#pragma unroll
    for (int j = 0; j < 4; ++j) {
      int col = (wave * 4 + j) * 16 + lrow;
      float bias = B1[n * FF + fbase + col];
#pragma unroll
      for (int r = 0; r < 8; ++r) {
        float v = acc1[j][r] + bias;
        hs[(r + hi16 * 8) * 512 + col] = (__bf16)(v > 0.f ? v : 0.f);
      }
    }

    // ---- GEMM2 partial: acc2 += h_chunk @ W2[fbase:fbase+512, :] ----
    {
      int buf = 0;
      stage_w2(0, 0);
      for (int k0 = 0; k0 < 512; k0 += 32) {
        if (k0 + 32 < 512) { stage_w2(buf ^ 1, k0 + 32); wait_async8(); }
        else               { wait_async0(); }
        __syncthreads();
        v16bf a = ld_frag_a(&hs[lrow * 512 + k0 + hi16 * 8]);
#pragma unroll
        for (int j = 0; j < 4; ++j) {
          int col = (wave * 4 + j) * 16 + lrow;
          v16bf b = ld_frag_b(&wst[buf][col * 32 + hi16 * 8]);
          acc2[j] = __builtin_amdgcn_wmma_f32_16x16x32_bf16(
              false, a, false, b, (short)0, acc2[j], false, false);
        }
        __syncthreads();
        buf ^= 1;
      }
    }
  }

  // ---- bias + scatter-store final 16 x 512 tile ----
#pragma unroll
  for (int j = 0; j < 4; ++j) {
    int col = (wave * 4 + j) * 16 + lrow;
    float b2 = B2[n * HH + col];
#pragma unroll
    for (int r = 0; r < 8; ++r) {
      int row = r + hi16 * 8;
      if (row < rows) {
        int tok = tl[row];
        out[(size_t)tok * HH + col] = acc2[j][r] + b2;
      }
    }
  }
}

extern "C" void kernel_launch(void* const* d_in, const int* in_sizes, int n_in,
                              void* d_out, int out_size, void* d_ws, size_t ws_size,
                              hipStream_t stream) {
  const float* x    = (const float*)d_in[0];
  const int*   bseq = (const int*)d_in[1];
  const float* W1   = (const float*)d_in[2];
  const float* B1   = (const float*)d_in[3];
  const float* W2   = (const float*)d_in[4];
  const float* B2   = (const float*)d_in[5];
  float* out = (float*)d_out;

  // ws layout (assumes ws_size >= ~34MB; fully rewritten every call)
  char* wsb = (char*)d_ws;
  int*    cnt = (int*)wsb;                                  // 32 B
  int*    idx = (int*)(wsb + 256);                          // 4*16384*4 = 256KB
  __bf16* xbf = (__bf16*)(wsb + 512 * 1024);                // 16MB
  __bf16* w1t = (__bf16*)(wsb + 512 * 1024 + 16777216);     // 8MB
  __bf16* w2t = (__bf16*)(wsb + 512 * 1024 + 16777216 + 8388608);  // 8MB

  const int xelems = NTOK * HH;  // 8,388,608
  zero_kernel<<<(out_size / 4 + 255) / 256, 256, 0, stream>>>(out, out_size, cnt);
  bucket_kernel<<<(NTOK + 255) / 256, 256, 0, stream>>>(bseq, cnt, idx);
  cvt_kernel<<<(xelems / 4 + 255) / 256, 256, 0, stream>>>(x, xbf, xelems);
  // W1 [n][H][F] -> w1t [n][F][H]
  transpose_cvt_kernel<<<dim3(FF / 32, HH / 32, NB), dim3(32, 8), 0, stream>>>(
      W1, w1t, HH, FF);
  // W2 [n][F][H] -> w2t [n][H][F]
  transpose_cvt_kernel<<<dim3(HH / 32, FF / 32, NB), dim3(32, 8), 0, stream>>>(
      W2, w2t, FF, HH);
  ffn_kernel<<<NB * MAXTILES, 256, 0, stream>>>(xbf, w1t, w2t, B1, B2, out, cnt,
                                                idx);
}
